// SoftBlobGIN_17617955848279
// MI455X (gfx1250) — matrix operand order repaired and checked by reading, not compiled, not run on hardware
//
#include <hip/hip_runtime.h>
#include <hip/hip_bf16.h>

typedef __attribute__((ext_vector_type(16))) _Float16 v16h;
typedef __attribute__((ext_vector_type(8)))  float    v8f;
typedef __attribute__((ext_vector_type(4)))  float    v4f;

#define HD   128
#define IND  64
#define KB   8     // blobs
#define CC   10
#define GG   256
#define LL   3
#define ROWT 4     // row tiles per block in WMMA gemm

// ---------------------------------------------------------------------------
// Pack W[Kd,Nd] (row-major f32) into f16 WMMA B-fragment order:
//   P[(ct*(Kd/32)+kt)*512 + lane*16 + e] = W[(kt*32 + (lane>>4)*8 + (e&7) + (e>>3)*16)*Nd + ct*16 + (lane&15)]
// so a wave's B fragment is one contiguous 32-byte load per (ct,kt).
// ---------------------------------------------------------------------------
__global__ void pack_w_kernel(const float* __restrict__ W, _Float16* __restrict__ P,
                              int Kd, int Nd) {
  const long long n = (long long)Kd * Nd;
  long long i = (long long)blockIdx.x * blockDim.x + threadIdx.x;
  if (i >= n) return;
  const int e    = (int)(i & 15);
  const int lane = (int)((i >> 4) & 31);
  const long long rest = i >> 9;          // ct*(Kd/32)+kt
  const int nkt = Kd >> 5;
  const int kt = (int)(rest % nkt);
  const int ct = (int)(rest / nkt);
  const int k   = kt * 32 + (lane >> 4) * 8 + (e & 7) + ((e >> 3) << 4);
  const int col = ct * 16 + (lane & 15);
  P[i] = (_Float16)W[(long long)k * Nd + col];
}

// ---------------------------------------------------------------------------
// WMMA GEMM:  C[M,Nd] = (A [+ A2]) [M,Kd] @ W [Kd,Nd] + bias
// W supplied pre-packed (f16 fragment order). block = 32*(Nd/16) threads,
// grid.x tiles M by 16*ROWT. Kd % 32 == 0, Nd % 16 == 0, rows 32B-aligned.
// Out-of-range rows are clamped on load (never stored), keeping the inner
// loop free of exec-mask manipulation.
// ---------------------------------------------------------------------------
__global__ void wmma_gemm_kernel(const float* __restrict__ A,
                                 const float* __restrict__ A2,
                                 const _Float16* __restrict__ P,
                                 const float* __restrict__ bias,
                                 float* __restrict__ C,
                                 int M, int Kd, int Nd) {
  const int wave = threadIdx.x >> 5;
  const int lane = threadIdx.x & 31;
  const int col0 = wave * 16;
  if (col0 >= Nd) return;                 // wave-uniform
  const int half = lane >> 4;             // 0 or 1
  const int l16  = lane & 15;
  const int rowbase = blockIdx.x * (16 * ROWT);
  const int nkt = Kd >> 5;

  // clamped per-tile row pointers (hoisted out of the K loop)
  const float* Arow[ROWT];
  const float* A2row[ROWT];
#pragma unroll
  for (int rt = 0; rt < ROWT; ++rt) {
    int arow = rowbase + rt * 16 + l16;
    arow = arow < M ? arow : (M - 1);
    Arow[rt]  = A + (long long)arow * Kd;
    A2row[rt] = A2 ? (A2 + (long long)arow * Kd) : nullptr;
  }

  v8f acc[ROWT];
#pragma unroll
  for (int rt = 0; rt < ROWT; ++rt) acc[rt] = (v8f){0.f,0.f,0.f,0.f,0.f,0.f,0.f,0.f};

  for (int kk = 0; kk < Kd; kk += 32) {
    // --- B fragment: one contiguous 32-byte load from packed weights ---
    const _Float16* Pw = P + ((((long long)(col0 >> 4)) * nkt + (kk >> 5)) << 9)
                           + ((long long)lane << 4);
    const v16h bfrag = *(const v16h*)Pw;

#pragma unroll
    for (int rt = 0; rt < ROWT; ++rt) {
      // two contiguous 8-float spans: kk+half*8 and kk+16+half*8
      const float* p0 = Arow[rt] + kk + half * 8;
      v4f a0 = *(const v4f*)(p0);
      v4f a1 = *(const v4f*)(p0 + 4);
      v4f a2 = *(const v4f*)(p0 + 16);
      v4f a3 = *(const v4f*)(p0 + 20);
      if (A2) {
        const float* q0 = A2row[rt] + kk + half * 8;
        a0 += *(const v4f*)(q0);
        a1 += *(const v4f*)(q0 + 4);
        a2 += *(const v4f*)(q0 + 16);
        a3 += *(const v4f*)(q0 + 20);
      }
      v16h afrag;
#pragma unroll
      for (int j = 0; j < 4; ++j) {
        afrag[j]      = (_Float16)a0[j];
        afrag[4 + j]  = (_Float16)a1[j];
        afrag[8 + j]  = (_Float16)a2[j];
        afrag[12 + j] = (_Float16)a3[j];
      }
      acc[rt] = __builtin_amdgcn_wmma_f32_16x16x32_f16(
          false, afrag, false, bfrag, (short)0, acc[rt], false, false);
    }
  }

  // --- store: element v -> row rowbase + rt*16 + half*8 + v, col col0+l16 ---
  const int ccol = col0 + l16;
  const float bi = bias ? bias[ccol] : 0.f;
#pragma unroll
  for (int rt = 0; rt < ROWT; ++rt) {
#pragma unroll
    for (int v = 0; v < 8; ++v) {
      const int crow = rowbase + rt * 16 + (half << 3) + v;
      if (crow < M) C[(long long)crow * Nd + ccol] = acc[rt][v] + bi;
    }
  }
}

// ---------------------------------------------------------------------------
// zero fill
// ---------------------------------------------------------------------------
__global__ void zero_kernel(float* __restrict__ p, long long n) {
  long long i = (long long)blockIdx.x * blockDim.x + threadIdx.x;
  const long long stride = (long long)gridDim.x * blockDim.x;
  for (; i < n; i += stride) p[i] = 0.f;
}

// ---------------------------------------------------------------------------
// edge scatter: aggr[dst] += relu(h[src] + elin_w + elin_b), 4 feats / thread
// ---------------------------------------------------------------------------
__global__ void edge_scatter_kernel(const float* __restrict__ h,
                                    const int* __restrict__ src,
                                    const int* __restrict__ dst,
                                    const float* __restrict__ ew,
                                    const float* __restrict__ eb,
                                    float* __restrict__ aggr, int E) {
  const long long t = (long long)blockIdx.x * blockDim.x + threadIdx.x;
  const int per = HD / 4;                           // 32 chunks
  const long long edge = t / per;
  if (edge >= E) return;
  const int f0 = (int)(t % per) * 4;
  const int s = src[edge], d = dst[edge];
  const v4f hv = *(const v4f*)(h + (long long)s * HD + f0);
  const v4f ev = *(const v4f*)(ew + f0);
  const v4f bv = *(const v4f*)(eb + f0);
  float* ad = aggr + (long long)d * HD + f0;
#pragma unroll
  for (int i = 0; i < 4; ++i) {
    float m = hv[i] + ev[i] + bv[i];
    m = m > 0.f ? m : 0.f;
    atomicAdd(ad + i, m);
  }
}

// ---------------------------------------------------------------------------
// per-column batch stats: mean[c], rstd[c] = rsqrt(var+eps)
// ---------------------------------------------------------------------------
__global__ void colstats_kernel(const float* __restrict__ X, int M, int Hd,
                                float* __restrict__ mean, float* __restrict__ rstd) {
  const int c = blockIdx.x;
  float s = 0.f, s2 = 0.f;
  for (int r = threadIdx.x; r < M; r += blockDim.x) {
    const float v = X[(long long)r * Hd + c];
    s += v; s2 += v * v;
  }
  __shared__ float sh[256], sh2[256];
  sh[threadIdx.x] = s; sh2[threadIdx.x] = s2;
  __syncthreads();
  for (int o = blockDim.x >> 1; o > 0; o >>= 1) {
    if ((int)threadIdx.x < o) {
      sh[threadIdx.x]  += sh[threadIdx.x + o];
      sh2[threadIdx.x] += sh2[threadIdx.x + o];
    }
    __syncthreads();
  }
  if (threadIdx.x == 0) {
    const float m = sh[0] / (float)M;
    const float v = sh2[0] / (float)M - m * m;
    mean[c] = m;
    rstd[c] = rsqrtf(v + 1e-5f);
  }
}

__global__ void bnapply_kernel(const float* __restrict__ X, float* __restrict__ Y,
                               const float* __restrict__ mean, const float* __restrict__ rstd,
                               const float* __restrict__ g, const float* __restrict__ b,
                               long long n, int Hd, int relu) {
  long long i = (long long)blockIdx.x * blockDim.x + threadIdx.x;
  const long long stride = (long long)gridDim.x * blockDim.x;
  for (; i < n; i += stride) {
    const int c = (int)(i % Hd);
    float v = (X[i] - mean[c]) * rstd[c] * g[c] + b[c];
    if (relu && v < 0.f) v = 0.f;
    Y[i] = v;
  }
}

// ---------------------------------------------------------------------------
// blob-assignment head: q = relu(h@bh1+b1); logits = q@bh2+b2;
// assign = softmax(logits + gumbel)  (TAU = 1)
// ---------------------------------------------------------------------------
__global__ void assign_kernel(const float* __restrict__ h,
                              const float* __restrict__ bh1_w, const float* __restrict__ bh1_b,
                              const float* __restrict__ bh2_w, const float* __restrict__ bh2_b,
                              const float* __restrict__ gumbel,
                              float* __restrict__ assign, int N) {
  const int i = blockIdx.x * blockDim.x + threadIdx.x;
  if (i >= N) return;
  const float* hr = h + (long long)i * HD;
  float logit[KB];
#pragma unroll
  for (int k = 0; k < KB; ++k) logit[k] = bh2_b[k];
  for (int j = 0; j < HD / 2; ++j) {
    float q = bh1_b[j];
    for (int k = 0; k < HD; ++k) q += hr[k] * bh1_w[k * (HD / 2) + j];
    q = q > 0.f ? q : 0.f;
#pragma unroll
    for (int k = 0; k < KB; ++k) logit[k] += q * bh2_w[j * KB + k];
  }
  float mx = -1e30f;
#pragma unroll
  for (int k = 0; k < KB; ++k) {
    logit[k] += gumbel[(long long)i * KB + k];
    mx = fmaxf(mx, logit[k]);
  }
  float ssum = 0.f;
#pragma unroll
  for (int k = 0; k < KB; ++k) { logit[k] = __expf(logit[k] - mx); ssum += logit[k]; }
  const float inv = 1.f / ssum;
#pragma unroll
  for (int k = 0; k < KB; ++k) assign[(long long)i * KB + k] = logit[k] * inv;
}

// ---------------------------------------------------------------------------
// per-graph pooling accumulators (block = 1 node, 128 threads)
// ---------------------------------------------------------------------------
__global__ void pool_kernel(const float* __restrict__ h, const float* __restrict__ assign,
                            const int* __restrict__ batch,
                            float* __restrict__ gsum, float* __restrict__ counts,
                            float* __restrict__ den, float* __restrict__ num, int N) {
  const int i = blockIdx.x;
  const int f = threadIdx.x;
  const int g = batch[i];
  const float hv = h[(long long)i * HD + f];
  atomicAdd(&gsum[g * HD + f], hv);
  __shared__ float as[KB];
  if (f < KB) {
    const float a = assign[(long long)i * KB + f];
    as[f] = a;
    atomicAdd(&den[g * KB + f], a);
  }
  if (f == 0) atomicAdd(&counts[g], 1.0f);
  __syncthreads();
#pragma unroll
  for (int k = 0; k < KB; ++k)
    atomicAdd(&num[((long long)g * KB + k) * HD + f], as[k] * hv);
}

__global__ void blobdiv_kernel(const float* __restrict__ num, const float* __restrict__ den,
                               float* __restrict__ blobs, long long n) {
  long long i = (long long)blockIdx.x * blockDim.x + threadIdx.x;
  if (i >= n) return;
  blobs[i] = num[i] / (den[i / HD] + 1e-8f);
}

// relu then layernorm over each 128-wide row (block = 1 row)
__global__ void relu_ln_kernel(float* __restrict__ X, const float* __restrict__ g,
                               const float* __restrict__ b) {
  const int r = blockIdx.x;
  const int f = threadIdx.x;  // 128
  float v = X[(long long)r * HD + f];
  v = v > 0.f ? v : 0.f;
  __shared__ float sh[HD], sh2[HD];
  sh[f] = v; sh2[f] = v * v;
  __syncthreads();
  for (int o = HD >> 1; o > 0; o >>= 1) {
    if (f < o) { sh[f] += sh[f + o]; sh2[f] += sh2[f + o]; }
    __syncthreads();
  }
  const float m  = sh[0] / (float)HD;
  const float va = sh2[0] / (float)HD - m * m;
  X[(long long)r * HD + f] = (v - m) * rsqrtf(va + 1e-5f) * g[f] + b[f];
}

__global__ void maxk_kernel(const float* __restrict__ X, float* __restrict__ out, int n) {
  const int i = blockIdx.x * blockDim.x + threadIdx.x;  // G*HD
  if (i >= n) return;
  const int g = i / HD, f = i % HD;
  float m = -1e30f;
#pragma unroll
  for (int k = 0; k < KB; ++k)
    m = fmaxf(m, X[((long long)g * KB + k) * HD + f]);
  out[i] = m;
}

__global__ void concat_kernel(const float* __restrict__ gsum, const float* __restrict__ counts,
                              const float* __restrict__ bemb, float* __restrict__ z, int n) {
  const int i = blockIdx.x * blockDim.x + threadIdx.x;  // G*2H
  if (i >= n) return;
  const int g = i / (2 * HD), j = i % (2 * HD);
  z[i] = (j < HD) ? gsum[g * HD + j] / counts[g] : bemb[g * HD + (j - HD)];
}

__global__ void clf2_kernel(const float* __restrict__ c1, const float* __restrict__ w,
                            const float* __restrict__ b, float* __restrict__ out, int n) {
  const int i = blockIdx.x * blockDim.x + threadIdx.x;  // G*C
  if (i >= n) return;
  const int g = i / CC, c = i % CC;
  float s = b[c];
  for (int k = 0; k < HD; ++k) s += c1[(long long)g * HD + k] * w[k * CC + c];
  out[i] = s;
}

// ---------------------------------------------------------------------------
// host orchestration
// ---------------------------------------------------------------------------
static inline int cdiv(long long a, long long b) { return (int)((a + b - 1) / b); }

extern "C" void kernel_launch(void* const* d_in, const int* in_sizes, int n_in,
                              void* d_out, int out_size, void* d_ws, size_t ws_size,
                              hipStream_t stream) {
  const float* x        = (const float*)d_in[0];
  const int*   eidx     = (const int*)  d_in[1];
  const int*   batch    = (const int*)  d_in[2];
  const float* gumbel   = (const float*)d_in[3];
  const float* inp_w    = (const float*)d_in[4];
  const float* inp_b    = (const float*)d_in[5];
  const float* elin_w   = (const float*)d_in[6];
  const float* elin_b   = (const float*)d_in[7];
  const float* mlp1_w   = (const float*)d_in[8];
  const float* mlp1_b   = (const float*)d_in[9];
  const float* mlpbn_g  = (const float*)d_in[10];
  const float* mlpbn_b  = (const float*)d_in[11];
  const float* mlp2_w   = (const float*)d_in[12];
  const float* mlp2_b   = (const float*)d_in[13];
  const float* bn_g     = (const float*)d_in[14];
  const float* bn_b     = (const float*)d_in[15];
  const float* bh1_w    = (const float*)d_in[16];
  const float* bh1_b    = (const float*)d_in[17];
  const float* bh2_w    = (const float*)d_in[18];
  const float* bh2_b    = (const float*)d_in[19];
  const float* bm_w     = (const float*)d_in[20];
  const float* bm_b     = (const float*)d_in[21];
  const float* ln_g     = (const float*)d_in[22];
  const float* ln_b     = (const float*)d_in[23];
  const float* clf1_w   = (const float*)d_in[24];
  const float* clf1_b   = (const float*)d_in[25];
  const float* clfbn_g  = (const float*)d_in[26];
  const float* clfbn_b  = (const float*)d_in[27];
  const float* clf2_w   = (const float*)d_in[28];
  const float* clf2_b   = (const float*)d_in[29];

  const int N = in_sizes[0] / IND;
  const int E = in_sizes[1] / 2;
  const int* src = eidx;
  const int* dst = eidx + E;

  // --- carve workspace (floats first, then 32B-aligned f16 packed weights) ---
  const long long NH = (long long)N * HD;
  float* f = (float*)d_ws;
  float* h      = f; f += NH;
  float* aggr   = f; f += NH;          // also reused as t2
  float* t1     = f; f += NH;          // also reused as blob-gemm output (2048x128)
  float* assign = f; f += (long long)N * KB;
  float* gsum   = f; f += GG * HD;     // |
  float* counts = f; f += GG;          // | contiguous zero region
  float* den    = f; f += GG * KB;     // |
  float* num    = f; f += (long long)GG * KB * HD;  // |
  float* blobs  = f; f += (long long)GG * KB * HD;
  float* bemb   = f; f += GG * HD;
  float* zcat   = f; f += GG * 2 * HD;
  float* c1     = f; f += GG * HD;
  float* mean   = f; f += 256;
  float* rstd   = f; f += 256;
  // align to 32 bytes for v16h loads
  uintptr_t pa = ((uintptr_t)f + 31) & ~(uintptr_t)31;
  _Float16* p16 = (_Float16*)pa;
  _Float16* p_inp  = p16; p16 += IND * HD;                 // 8192
  _Float16* p_m1   = p16; p16 += (long long)LL * HD * HD;  // 49152
  _Float16* p_m2   = p16; p16 += (long long)LL * HD * HD;  // 49152
  _Float16* p_bm   = p16; p16 += HD * HD;                  // 16384
  _Float16* p_c1   = p16; p16 += 2 * HD * HD;              // 32768
  const size_t need = (size_t)((char*)p16 - (char*)d_ws);
  if (ws_size < need) return;

  const long long zeroPool = (long long)GG * HD + GG + GG * KB + (long long)GG * KB * HD;

  // 0) pack all GEMM weights to f16 fragment order
  pack_w_kernel<<<cdiv(IND * HD, 256), 256, 0, stream>>>(inp_w, p_inp, IND, HD);
  for (int l = 0; l < LL; ++l) {
    pack_w_kernel<<<cdiv(HD * HD, 256), 256, 0, stream>>>(
        mlp1_w + (long long)l * HD * HD, p_m1 + (long long)l * HD * HD, HD, HD);
    pack_w_kernel<<<cdiv(HD * HD, 256), 256, 0, stream>>>(
        mlp2_w + (long long)l * HD * HD, p_m2 + (long long)l * HD * HD, HD, HD);
  }
  pack_w_kernel<<<cdiv(HD * HD, 256), 256, 0, stream>>>(bm_w, p_bm, HD, HD);
  pack_w_kernel<<<cdiv(2 * HD * HD, 256), 256, 0, stream>>>(clf1_w, p_c1, 2 * HD, HD);

  // 1) h = x @ inp_w + inp_b           [N,64]@[64,128]
  wmma_gemm_kernel<<<cdiv(N, 16 * ROWT), 256, 0, stream>>>(
      x, nullptr, p_inp, inp_b, h, N, IND, HD);

  // 2) GINE layers
  for (int l = 0; l < LL; ++l) {
    zero_kernel<<<4096, 256, 0, stream>>>(aggr, NH);
    {
      const long long T = (long long)E * (HD / 4);
      edge_scatter_kernel<<<cdiv(T, 256), 256, 0, stream>>>(
          h, src, dst, elin_w + l * HD, elin_b + l * HD, aggr, E);
    }
    // t1 = (h + aggr) @ mlp1_w[l] + b
    wmma_gemm_kernel<<<cdiv(N, 16 * ROWT), 256, 0, stream>>>(
        h, aggr, p_m1 + (long long)l * HD * HD, mlp1_b + l * HD, t1, N, HD, HD);
    colstats_kernel<<<HD, 256, 0, stream>>>(t1, N, HD, mean, rstd);
    bnapply_kernel<<<4096, 256, 0, stream>>>(t1, t1, mean, rstd,
        mlpbn_g + l * HD, mlpbn_b + l * HD, NH, HD, 1);
    // t2(=aggr) = t1 @ mlp2_w[l] + b
    wmma_gemm_kernel<<<cdiv(N, 16 * ROWT), 256, 0, stream>>>(
        t1, nullptr, p_m2 + (long long)l * HD * HD, mlp2_b + l * HD, aggr, N, HD, HD);
    colstats_kernel<<<HD, 256, 0, stream>>>(aggr, N, HD, mean, rstd);
    bnapply_kernel<<<4096, 256, 0, stream>>>(aggr, h, mean, rstd,
        bn_g + l * HD, bn_b + l * HD, NH, HD, 1);
  }

  // 3) blob assignment
  assign_kernel<<<cdiv(N, 128), 128, 0, stream>>>(h, bh1_w, bh1_b, bh2_w, bh2_b,
                                                  gumbel, assign, N);

  // 4) per-graph pooling
  zero_kernel<<<1024, 256, 0, stream>>>(gsum, zeroPool);   // gsum|counts|den|num contiguous
  pool_kernel<<<N, HD, 0, stream>>>(h, assign, batch, gsum, counts, den, num, N);

  // 5) blobs = num / (den + 1e-8); r = relu(blobs @ bm_w + bm_b); layernorm; max over K
  {
    const long long nb = (long long)GG * KB * HD;
    blobdiv_kernel<<<cdiv(nb, 256), 256, 0, stream>>>(num, den, blobs, nb);
    const int Mb = GG * KB;  // 2048
    wmma_gemm_kernel<<<cdiv(Mb, 16 * ROWT), 256, 0, stream>>>(
        blobs, nullptr, p_bm, bm_b, t1, Mb, HD, HD);
    relu_ln_kernel<<<Mb, HD, 0, stream>>>(t1, ln_g, ln_b);
    maxk_kernel<<<cdiv(GG * HD, 256), 256, 0, stream>>>(t1, bemb, GG * HD);
  }

  // 6) classifier
  concat_kernel<<<cdiv(GG * 2 * HD, 256), 256, 0, stream>>>(gsum, counts, bemb, zcat,
                                                            GG * 2 * HD);
  wmma_gemm_kernel<<<cdiv(GG, 16 * ROWT), 256, 0, stream>>>(
      zcat, nullptr, p_c1, clf1_b, c1, GG, 2 * HD, HD);
  colstats_kernel<<<HD, 256, 0, stream>>>(c1, GG, HD, mean, rstd);
  bnapply_kernel<<<256, 256, 0, stream>>>(c1, c1, mean, rstd, clfbn_g, clfbn_b,
                                          (long long)GG * HD, HD, 1);
  clf2_kernel<<<cdiv(GG * CC, 256), 256, 0, stream>>>(c1, clf2_w, clf2_b,
                                                      (float*)d_out, GG * CC);
}
